// PointnetPP_13469017440296
// MI455X (gfx1250) — compile-verified, matrix operand order checked by reading, not run on hardware
//
#include <hip/hip_runtime.h>
#include <hip/hip_bf16.h>
#include <stdint.h>

// ---------------------------------------------------------------------------
// PointNet++ forward on gfx1250 (MI455X, wave32).
// All MLP layers run through v_wmma_f32_16x16x32_f16 GEMMs (f16 in, f32 acc;
// BN renormalizes immediately so f16 activations are the right precision).
// ~140 GFLOP of matrix work, activations L2-resident -> matrix-core bound.
// GEMM: 8-wave blocks, 128x64 block tile, 4 WMMA accumulators per wave,
// A tile staged into LDS via the Tensor Data Mover (TDM) when available.
// ---------------------------------------------------------------------------

typedef __attribute__((ext_vector_type(16))) _Float16 v16h;
typedef __attribute__((ext_vector_type(8)))  _Float16 h8;
typedef __attribute__((ext_vector_type(8)))  float    v8f;
typedef __attribute__((ext_vector_type(4)))  unsigned int u32x4;
typedef __attribute__((ext_vector_type(8)))  int      i32x8;
typedef __attribute__((ext_vector_type(4)))  int      i32x4;

#define BN_EPS 1e-5f

#if __has_builtin(__builtin_amdgcn_tensor_load_to_lds) && \
    __has_builtin(__builtin_amdgcn_s_wait_tensorcnt)
#define HAVE_TDM 1
#else
#define HAVE_TDM 0
#endif

#if HAVE_TDM
// Issue a TDM DMA of a [rows x 32] f16 tile (row stride Kp elements) from
// global `gptr` into LDS offset `lds_off`. D# layout per CDNA5 ISA ch.8.
__device__ __forceinline__ void tdm_load_a_tile(const _Float16* gptr, unsigned lds_off,
                                                int Kp, int rows)
{
  const unsigned long long ga = (unsigned long long)(uintptr_t)gptr;
  u32x4 g0 = {};
  g0[0] = 1u;                                   // count=1, user descriptor
  g0[1] = lds_off;                              // lds_addr (bytes)
  g0[2] = (unsigned)(ga & 0xFFFFFFFFu);         // global_addr[31:0]
  g0[3] = (unsigned)((ga >> 32) & 0x1FFFFFFu)   // global_addr[56:32]
        | (2u << 30);                           // type=2 ("image")
  const unsigned long long stride = (unsigned long long)Kp; // dim0 stride (elems)
  i32x8 g1 = {};
  g1[0] = 0x00010000;                           // data_size=1 -> 2 bytes
  g1[1] = (int)((Kp & 0xFFFF) << 16);           // tensor_dim0[15:0]
  g1[2] = (int)(((Kp >> 16) & 0xFFFF) | ((rows & 0xFFFF) << 16)); // dim0 hi | dim1 lo
  g1[3] = (int)(((rows >> 16) & 0xFFFF) | (32u << 16));           // dim1 hi | tile_dim0=32
  g1[4] = (int)(rows & 0xFFFF);                 // tile_dim1 = rows (tile_dim2=0)
  g1[5] = (int)(stride & 0xFFFFFFFFu);          // tensor_dim0_stride lo
  g1[6] = (int)((stride >> 32) & 0xFFFF);       // dim0_stride hi | dim1_stride lo(=0)
  g1[7] = 0;
  i32x4 z4 = {};
#if __clang_major__ >= 23
  i32x8 z8 = {};
  __builtin_amdgcn_tensor_load_to_lds(g0, g1, z4, z4, z8, 0);
#else
  __builtin_amdgcn_tensor_load_to_lds(g0, g1, z4, z4, 0);
#endif
}
#endif

// ------------------------- farthest point sampling -------------------------
__global__ __launch_bounds__(256) void fps_kernel(const float* __restrict__ pos,
                                                  int N, int ns, int* __restrict__ out)
{
  __shared__ float dist[4096];
  __shared__ float red[256];
  __shared__ int   redi[256];
  __shared__ int   cur;
  const int tid = threadIdx.x;
  const float* p = pos + (size_t)blockIdx.x * N * 3;
  const float x0 = p[0], y0 = p[1], z0 = p[2];
  for (int i = tid; i < N; i += 256) {
    float dx = p[i*3]-x0, dy = p[i*3+1]-y0, dz = p[i*3+2]-z0;
    dist[i] = dx*dx + dy*dy + dz*dz;
  }
  if (tid == 0) out[(size_t)blockIdx.x*ns] = 0;
  __syncthreads();
  for (int s = 1; s < ns; ++s) {
    float bm = -1.f; int bi = 0;
    for (int i = tid; i < N; i += 256) { float d = dist[i]; if (d > bm) { bm = d; bi = i; } }
    red[tid] = bm; redi[tid] = bi;
    __syncthreads();
    for (int off = 128; off > 0; off >>= 1) {
      if (tid < off && red[tid+off] > red[tid]) { red[tid] = red[tid+off]; redi[tid] = redi[tid+off]; }
      __syncthreads();
    }
    if (tid == 0) { cur = redi[0]; out[(size_t)blockIdx.x*ns + s] = cur; }
    __syncthreads();
    const int c = cur;
    const float cx = p[c*3], cy = p[c*3+1], cz = p[c*3+2];
    for (int i = tid; i < N; i += 256) {
      float dx = p[i*3]-cx, dy = p[i*3+1]-cy, dz = p[i*3+2]-cz;
      float nd = dx*dx + dy*dy + dz*dz;
      if (nd < dist[i]) dist[i] = nd;
    }
    __syncthreads();
  }
}

// ------------------------------ kNN (top-K min) ----------------------------
__global__ __launch_bounds__(256) void knn_kernel(const float* __restrict__ pos,
                                                  const int* __restrict__ sidx,
                                                  int N, int ns, int K, int* __restrict__ out)
{
  __shared__ float dist[4096];
  __shared__ float red[256];
  __shared__ int   redi[256];
  const int tid = threadIdx.x;
  const int b = blockIdx.x / ns, s = blockIdx.x % ns;
  const float* p = pos + (size_t)b * N * 3;
  const int q = sidx[b*ns + s];
  const float qx = p[q*3], qy = p[q*3+1], qz = p[q*3+2];
  for (int i = tid; i < N; i += 256) {
    float dx = p[i*3]-qx, dy = p[i*3+1]-qy, dz = p[i*3+2]-qz;
    dist[i] = dx*dx + dy*dy + dz*dz;
  }
  __syncthreads();
  for (int t = 0; t < K; ++t) {
    float bm = 3.0e38f; int bi = 0;
    for (int i = tid; i < N; i += 256) { float d = dist[i]; if (d < bm) { bm = d; bi = i; } }
    red[tid] = bm; redi[tid] = bi;
    __syncthreads();
    for (int off = 128; off > 0; off >>= 1) {
      if (tid < off && red[tid+off] < red[tid]) { red[tid] = red[tid+off]; redi[tid] = redi[tid+off]; }
      __syncthreads();
    }
    if (tid == 0) { out[(size_t)blockIdx.x*K + t] = redi[0]; dist[redi[0]] = 3.0e38f; }
    __syncthreads();
  }
}

// ------------------------------- small helpers -----------------------------
__global__ void gather_pos_kernel(const float* __restrict__ pos, const int* __restrict__ sidx,
                                  int N, int ns, int Btot, float* __restrict__ out)
{
  int t = blockIdx.x*blockDim.x + threadIdx.x;
  if (t >= Btot*ns*3) return;
  int c = t % 3; int s = (t/3) % ns; int b = t/(3*ns);
  out[t] = pos[((size_t)b*N + sidx[b*ns+s])*3 + c];
}

__global__ void zero2_kernel(float* a, float* b, int n)
{
  int t = blockIdx.x*blockDim.x + threadIdx.x;
  if (t < n) { a[t] = 0.f; b[t] = 0.f; }
}

__global__ void wpack_kernel(const float* __restrict__ Wsrc, int Cin, int Kp, int Cout,
                             _Float16* __restrict__ out)
{
  int t = blockIdx.x*blockDim.x + threadIdx.x;
  if (t >= Cout*Kp) return;
  int c = t % Kp, o = t / Kp;
  out[t] = (_Float16)(c < Cin ? Wsrc[(size_t)o*Cin + c] : 0.f);
}

// grouped feature builder: [gp(3) | x[nidx](Cx) | zero-pad] -> f16 [B*ns*K, Kp]
__global__ void group_pack_kernel(const float* __restrict__ pos, const float* __restrict__ x,
                                  const int* __restrict__ sidx, const int* __restrict__ nidx,
                                  int N, int ns, int K, int Cx, int Kp, int Btot,
                                  _Float16* __restrict__ out)
{
  size_t i = (size_t)blockIdx.x*blockDim.x + threadIdx.x;
  size_t total = (size_t)Btot*ns*K*Kp;
  if (i >= total) return;
  int c = (int)(i % Kp);
  size_t row = i / Kp;
  int k = (int)(row % K);
  int s = (int)((row / K) % ns);
  int b = (int)(row / ((size_t)K*ns));
  int g = b*ns + s;
  float v = 0.f;
  if (c < 3) {
    int j = nidx[(size_t)g*K + k];
    int q = sidx[g];
    v = pos[((size_t)b*N + j)*3 + c] - pos[((size_t)b*N + q)*3 + c];
  } else if (c < 3 + Cx) {
    int j = nidx[(size_t)g*K + k];
    v = x[((size_t)b*N + j)*(size_t)Cx + (c-3)];
  }
  out[i] = (_Float16)v;
}

__global__ void pack2_kernel(const float* __restrict__ A, int Ca,
                             const float* __restrict__ Bs, int Cb,
                             int Kp, size_t R, _Float16* __restrict__ out)
{
  size_t i = (size_t)blockIdx.x*blockDim.x + threadIdx.x;
  if (i >= R*(size_t)Kp) return;
  int c = (int)(i % Kp); size_t r = i / Kp;
  float v = 0.f;
  if (c < Ca)            v = A[r*(size_t)Ca + c];
  else if (c < Ca + Cb)  v = Bs[r*(size_t)Cb + (c - Ca)];
  out[i] = (_Float16)v;
}

// 3-NN inverse-distance interpolation indices + weights
__global__ void interp_idx_kernel(const float* __restrict__ p1, const float* __restrict__ p2,
                                  int M, int Np, int Btot,
                                  int* __restrict__ idx3, float* __restrict__ w3)
{
  int t = blockIdx.x*blockDim.x + threadIdx.x;
  if (t >= Btot*Np) return;
  int b = t / Np, i = t % Np;
  const float* q  = p2 + ((size_t)b*Np + i)*3;
  const float* pp = p1 + (size_t)b*M*3;
  float bd[3] = {3.0e38f, 3.0e38f, 3.0e38f};
  int   bi[3] = {0, 0, 0};
  for (int j = 0; j < M; ++j) {
    float dx = pp[j*3]-q[0], dy = pp[j*3+1]-q[1], dz = pp[j*3+2]-q[2];
    float d = dx*dx + dy*dy + dz*dz;
    if (d < bd[0])      { bd[2]=bd[1]; bi[2]=bi[1]; bd[1]=bd[0]; bi[1]=bi[0]; bd[0]=d; bi[0]=j; }
    else if (d < bd[1]) { bd[2]=bd[1]; bi[2]=bi[1]; bd[1]=d; bi[1]=j; }
    else if (d < bd[2]) { bd[2]=d; bi[2]=j; }
  }
  int k = M < 3 ? M : 3;
  float r[3] = {0.f,0.f,0.f}, rs = 0.f;
  for (int j = 0; j < k; ++j) {
    float d = bd[j] > 0.f ? sqrtf(bd[j]) : 0.f;
    r[j] = 1.f / (d + 1e-8f); rs += r[j];
  }
  for (int j = 0; j < 3; ++j) {
    idx3[(size_t)t*3 + j] = j < k ? bi[j] : 0;
    w3[(size_t)t*3 + j]   = j < k ? r[j]/rs : 0.f;
  }
}

// up-path packer: [interp(Cf) | skipA(Ca) | skipB(Cb) | zero-pad] -> f16 [B*Np, Kp]
__global__ void up_pack_kernel(const float* __restrict__ feat, const int* __restrict__ idx3,
                               const float* __restrict__ w3,
                               const float* __restrict__ skipA, const float* __restrict__ skipB,
                               int M, int Np, int Cf, int Ca, int Cb, int Kp, int Btot,
                               _Float16* __restrict__ out)
{
  size_t i = (size_t)blockIdx.x*blockDim.x + threadIdx.x;
  size_t total = (size_t)Btot*Np*Kp;
  if (i >= total) return;
  int c = (int)(i % Kp);
  size_t row = i / Kp;
  int b = (int)(row / Np);
  float v = 0.f;
  if (c < Cf) {
    const int*   id = idx3 + row*3;
    const float* ww = w3   + row*3;
    const float* fb = feat + (size_t)b*M*(size_t)Cf + c;
    v = ww[0]*fb[(size_t)id[0]*Cf] + ww[1]*fb[(size_t)id[1]*Cf] + ww[2]*fb[(size_t)id[2]*Cf];
  } else if (c < Cf + Ca) {
    v = skipA[row*(size_t)Ca + (c - Cf)];
  } else if (skipB && c < Cf + Ca + Cb) {
    v = skipB[row*(size_t)Cb + (c - Cf - Ca)];
  }
  out[i] = (_Float16)v;
}

// --------------------- fused WMMA GEMM + bias + BN stats -------------------
// Y[r,o] = sum_k X[r,k]*Wt[o,k] + bias[o], with per-channel sum/sumsq
// accumulated via f32 global atomics for the BN pass.
// Block = 256 threads = 8 waves. Block tile = 128 rows x 64 cols.
// Wave w owns rows [w*16, w*16+16) and ALL 64 cols -> 4 accumulators; one A
// fragment is reused across 4 WMMAs per k-step. A tile (128x32, 8KB) is DMA'd
// into LDS by the Tensor Data Mover; W tile (64x32, 4KB) staged by the block.
// Caller guarantees R%128==0, Cout%64==0, Kp%32==0 -> no guards, EXEC all 1s.
__global__ __launch_bounds__(256)
void wmma_gemm_kernel(const _Float16* __restrict__ X,  // [R, Kp]
                      const _Float16* __restrict__ Wt, // [Cout, Kp]
                      const float* __restrict__ bias,  // [Cout]
                      float* __restrict__ Y,           // [R, Cout]
                      float* __restrict__ sum, float* __restrict__ sumsq,
                      int Kp, int Cout)
{
  __shared__ __align__(16) _Float16 As[128*32];  // A tile, row-major
  __shared__ __align__(16) _Float16 Bs[64*32];   // W tile, row(=out-chan)-major
  const int tid  = threadIdx.x;
  const int wave = tid >> 5;
  const int lane = tid & 31;
  const int m    = lane & 15;     // A-row / B-col within 16-tile for this lane
  const int half = lane >> 4;     // selects K-group per ISA f16 fragment layout
  const size_t r0 = (size_t)blockIdx.x * 128;
  const int oblk  = blockIdx.y * 64;

  const _Float16* Xrow = X + r0 * Kp;
  // W-tile stage mapping: thread copies 8 halfs (16B)
  const int wcol = tid >> 2, wkc = (tid & 3) * 8;
#if !HAVE_TDM
  const int ae0 = tid * 8, ae1 = (tid + 256) * 8;  // manual A-stage fallback
#endif

  v8f acc[4] = {};
  for (int k0 = 0; k0 < Kp; k0 += 32) {
#if HAVE_TDM
    if (wave == 0) {
      tdm_load_a_tile(Xrow + k0, (unsigned)(uintptr_t)(void*)As, Kp, 128);
      __builtin_amdgcn_s_wait_tensorcnt(0);
    }
#else
    *(h8*)&As[ae0] = *(const h8*)(Xrow + (size_t)(ae0 >> 5) * Kp + k0 + (ae0 & 31));
    *(h8*)&As[ae1] = *(const h8*)(Xrow + (size_t)(ae1 >> 5) * Kp + k0 + (ae1 & 31));
#endif
    // stage W tile (64 rows x 32): 256 threads x 8 halfs
    *(h8*)&Bs[wcol*32 + wkc] = *(const h8*)(Wt + (size_t)(oblk + wcol) * Kp + k0 + wkc);
    __builtin_prefetch((const void*)(Wt + (size_t)(oblk + wcol) * Kp + k0 + wkc + 32), 0, 3);
    __syncthreads();

    // A fragment (rows wave*16 + m): two contiguous 8-half chunks per ISA layout
    v16h a;
    {
      const _Float16* ap = &As[(wave*16 + m)*32];
      h8 lo = *(const h8*)(ap + 8*half);
      h8 hi = *(const h8*)(ap + 16 + 8*half);
#pragma unroll
      for (int e = 0; e < 8; ++e) { a[e] = lo[e]; a[8+e] = hi[e]; }
    }
#pragma unroll
    for (int t = 0; t < 4; ++t) {
      v16h bf;
      const _Float16* bp = &Bs[(t*16 + m)*32];
      h8 lo = *(const h8*)(bp + 8*half);
      h8 hi = *(const h8*)(bp + 16 + 8*half);
#pragma unroll
      for (int e = 0; e < 8; ++e) { bf[e] = lo[e]; bf[8+e] = hi[e]; }
      acc[t] = __builtin_amdgcn_wmma_f32_16x16x32_f16(false, a, false, bf,
                                                      (short)0, acc[t], false, false);
    }
    __syncthreads();
  }

  // C/D layout: lane holds column (t*16+m); VGPR r holds row (r + 8*half)
#pragma unroll
  for (int t = 0; t < 4; ++t) {
    const int o = oblk + t*16 + m;
    const float bb = bias[o];
    float s = 0.f, s2 = 0.f;
#pragma unroll
    for (int r = 0; r < 8; ++r) {
      float y = acc[t][r] + bb;
      Y[(r0 + wave*16 + r + 8*(size_t)half) * (size_t)Cout + o] = y;
      s += y; s2 += y*y;
    }
    atomicAdd(&sum[o],   s);
    atomicAdd(&sumsq[o], s2);
  }
}

// BN (training-mode batch stats) + ReLU; writes f16 (zero-padded to Kpn) for
// the next GEMM and/or f32 for pooling / output.
__global__ void bn_relu_kernel(const float* __restrict__ Y,
                               const float* __restrict__ sum, const float* __restrict__ sumsq,
                               float invR, int C, int Kpn, size_t R,
                               _Float16* __restrict__ out16, float* __restrict__ out32)
{
  size_t i = (size_t)blockIdx.x*blockDim.x + threadIdx.x;
  if (i >= R*(size_t)Kpn) return;
  int c = (int)(i % Kpn); size_t r = i / Kpn;
  if (c < C) {
    float mu  = sum[c] * invR;
    float var = sumsq[c] * invR - mu*mu;
    float h = fmaxf((Y[r*(size_t)C + c] - mu) * rsqrtf(var + BN_EPS), 0.f);
    if (out16) out16[i] = (_Float16)h;
    if (out32) out32[r*(size_t)C + c] = h;
  } else if (out16) {
    out16[i] = (_Float16)0.f;
  }
}

__global__ void maxpool_kernel(const float* __restrict__ in, int K, int C, size_t G,
                               float* __restrict__ out)
{
  size_t t = (size_t)blockIdx.x*blockDim.x + threadIdx.x;
  if (t >= G*(size_t)C) return;
  size_t g = t / C; int c = (int)(t % C);
  const float* p = in + g*(size_t)K*C + c;
  float m = -3.0e38f;
  for (int k = 0; k < K; ++k) m = fmaxf(m, p[(size_t)k*C]);
  out[t] = m;
}

// ---------------------------------------------------------------------------
extern "C" void kernel_launch(void* const* d_in, const int* in_sizes, int n_in,
                              void* d_out, int out_size, void* d_ws, size_t ws_size,
                              hipStream_t stream)
{
  (void)in_sizes; (void)n_in; (void)out_size; (void)ws_size;
  const int B = 16, N = 4096, K = 64;

  const float* x0  = (const float*)d_in[0];
  const float* pos = (const float*)d_in[1];
  const float* Wl[16]; const float* bl[16];
  { int wi = 2; for (int l = 0; l < 16; ++l) { Wl[l] = (const float*)d_in[wi++]; bl[l] = (const float*)d_in[wi++]; } }

  // --- workspace bump allocator ---
  char* wp = (char*)d_ws;
  auto alloc = [&](size_t bytes) { void* p = (void*)wp; wp += (bytes + 255) & ~(size_t)255; return p; };
  const size_t MAXE = 36ull*1024*1024;                 // max R*Kp over all layers
  _Float16* X16a = (_Float16*)alloc(MAXE*2);
  _Float16* X16b = (_Float16*)alloc(MAXE*2);
  float*    Y32  = (float*)   alloc(MAXE*4);
  _Float16* W16  = (_Float16*)alloc(720896ull*2);
  float* sum   = (float*)alloc(1024*4);
  float* sumsq = (float*)alloc(1024*4);
  int* fps1 = (int*)alloc((size_t)B*256*4);
  int* fps2 = (int*)alloc((size_t)B*128*4);
  int* knn1 = (int*)alloc((size_t)B*256*64*4);
  int* knn2 = (int*)alloc((size_t)B*128*64*4);
  float* pos1 = (float*)alloc((size_t)B*256*3*4);
  float* pos2 = (float*)alloc((size_t)B*128*3*4);
  float* xc1  = (float*)alloc((size_t)B*256*128*4);
  float* xc2  = (float*)alloc((size_t)B*128*256*4);
  float* xc3  = (float*)alloc((size_t)B*1024*4);
  float* ups1 = (float*)alloc((size_t)B*128*512*4);
  float* ups2 = (float*)alloc((size_t)B*256*512*4);
  int*   idx3 = (int*)  alloc((size_t)B*4096*3*4);
  float* w3   = (float*)alloc((size_t)B*4096*3*4);
  float* zeros3 = (float*)alloc(64*4);

  // one MLP layer: pack W -> zero stats -> WMMA GEMM -> BN+ReLU
  auto run_layer = [&](const float* W, const float* bias, int Cin, int Kp, int Cout,
                       size_t R, const _Float16* Xin, _Float16* Xout, int KpOut, float* out32) {
    int wtot = Cout * Kp;
    wpack_kernel<<<dim3((wtot+255)/256), dim3(256), 0, stream>>>(W, Cin, Kp, Cout, W16);
    zero2_kernel<<<dim3((Cout+255)/256), dim3(256), 0, stream>>>(sum, sumsq, Cout);
    dim3 g((unsigned)(R/128), (unsigned)(Cout/64));
    wmma_gemm_kernel<<<g, dim3(256), 0, stream>>>(Xin, W16, bias, Y32, sum, sumsq, Kp, Cout);
    int Kpn = Xout ? KpOut : Cout;
    size_t tot = R * (size_t)Kpn;
    bn_relu_kernel<<<dim3((unsigned)((tot+255)/256)), dim3(256), 0, stream>>>(
        Y32, sum, sumsq, 1.f/(float)R, Cout, Kpn, R, Xout, out32);
  };

  // =============== SA level 1: N=4096 -> ns=256, 6 -> [64,64,128] ===============
  fps_kernel<<<dim3(B), dim3(256), 0, stream>>>(pos, N, 256, fps1);
  gather_pos_kernel<<<dim3((B*256*3+255)/256), dim3(256), 0, stream>>>(pos, fps1, N, 256, B, pos1);
  knn_kernel<<<dim3(B*256), dim3(256), 0, stream>>>(pos, fps1, N, 256, K, knn1);
  {
    size_t R = (size_t)B*256*K;              // 262144 rows
    size_t tot = R*32;
    group_pack_kernel<<<dim3((unsigned)((tot+255)/256)), dim3(256), 0, stream>>>(
        pos, x0, fps1, knn1, N, 256, K, 3, 32, B, X16a);
    run_layer(Wl[0], bl[0],   6,  32,  64, R, X16a, X16b,  64, nullptr);
    run_layer(Wl[1], bl[1],  64,  64,  64, R, X16b, X16a,  64, nullptr);
    run_layer(Wl[2], bl[2],  64,  64, 128, R, X16a, nullptr, 0, Y32);   // in-place BN for pooling
    maxpool_kernel<<<dim3((unsigned)(((size_t)B*256*128+255)/256)), dim3(256), 0, stream>>>(
        Y32, K, 128, (size_t)B*256, xc1);
  }

  // =============== SA level 2: 256 -> ns=128, 131 -> [128,128,256] ==============
  fps_kernel<<<dim3(B), dim3(256), 0, stream>>>(pos1, 256, 128, fps2);
  gather_pos_kernel<<<dim3((B*128*3+255)/256), dim3(256), 0, stream>>>(pos1, fps2, 256, 128, B, pos2);
  knn_kernel<<<dim3(B*128), dim3(256), 0, stream>>>(pos1, fps2, 256, 128, K, knn2);
  {
    size_t R = (size_t)B*128*K;              // 131072 rows
    size_t tot = R*160;
    group_pack_kernel<<<dim3((unsigned)((tot+255)/256)), dim3(256), 0, stream>>>(
        pos1, xc1, fps2, knn2, 256, 128, K, 128, 160, B, X16a);
    run_layer(Wl[3], bl[3], 131, 160, 128, R, X16a, X16b, 128, nullptr);
    run_layer(Wl[4], bl[4], 128, 128, 128, R, X16b, X16a, 128, nullptr);
    run_layer(Wl[5], bl[5], 128, 128, 256, R, X16a, nullptr, 0, Y32);
    maxpool_kernel<<<dim3((unsigned)(((size_t)B*128*256+255)/256)), dim3(256), 0, stream>>>(
        Y32, K, 256, (size_t)B*128, xc2);
  }

  // =============== SA level 3 (global): 259 -> [256,512,1024], max over 128 =====
  {
    size_t R = (size_t)B*128;                // 2048 rows
    size_t tot = R*288;
    pack2_kernel<<<dim3((unsigned)((tot+255)/256)), dim3(256), 0, stream>>>(
        pos2, 3, xc2, 256, 288, R, X16a);
    run_layer(Wl[6], bl[6], 259, 288,  256, R, X16a, X16b, 256, nullptr);
    run_layer(Wl[7], bl[7], 256, 256,  512, R, X16b, X16a, 512, nullptr);
    run_layer(Wl[8], bl[8], 512, 512, 1024, R, X16a, nullptr, 0, Y32);
    maxpool_kernel<<<dim3((unsigned)(((size_t)B*1024+255)/256)), dim3(256), 0, stream>>>(
        Y32, 128, 1024, (size_t)B, xc3);
  }

  // =============== UP1: interp(1024,M=1)+skip(256) -> [512,512] =================
  zero2_kernel<<<dim3(1), dim3(64), 0, stream>>>(zeros3, zeros3, 48);  // global pos = 0
  interp_idx_kernel<<<dim3((B*128+255)/256), dim3(256), 0, stream>>>(zeros3, pos2, 1, 128, B, idx3, w3);
  {
    size_t R = (size_t)B*128; int Kp = 1280;
    up_pack_kernel<<<dim3((unsigned)((R*Kp+255)/256)), dim3(256), 0, stream>>>(
        xc3, idx3, w3, xc2, nullptr, 1, 128, 1024, 256, 0, Kp, B, X16a);
    run_layer(Wl[9],  bl[9],  1280, 1280, 512, R, X16a, X16b, 512, nullptr);
    run_layer(Wl[10], bl[10],  512,  512, 512, R, X16b, nullptr, 0, ups1);
  }

  // =============== UP2: interp(512,M=128)+skip(128) -> [512,512] ================
  interp_idx_kernel<<<dim3((B*256+255)/256), dim3(256), 0, stream>>>(pos2, pos1, 128, 256, B, idx3, w3);
  {
    size_t R = (size_t)B*256; int Kp = 640;
    up_pack_kernel<<<dim3((unsigned)((R*Kp+255)/256)), dim3(256), 0, stream>>>(
        ups1, idx3, w3, xc1, nullptr, 128, 256, 512, 128, 0, Kp, B, X16a);
    run_layer(Wl[11], bl[11], 640, 640, 512, R, X16a, X16b, 512, nullptr);
    run_layer(Wl[12], bl[12], 512, 512, 512, R, X16b, nullptr, 0, ups2);
  }

  // ========= UP3: interp(512,M=256)+skip(x0:3,pos:3) -> [512,512,512] ===========
  interp_idx_kernel<<<dim3((B*4096+255)/256), dim3(256), 0, stream>>>(pos1, pos, 256, 4096, B, idx3, w3);
  {
    size_t R = (size_t)B*4096; int Kp = 544;
    up_pack_kernel<<<dim3((unsigned)((R*Kp+255)/256)), dim3(256), 0, stream>>>(
        ups2, idx3, w3, x0, pos, 256, 4096, 512, 3, 3, Kp, B, X16a);
    run_layer(Wl[13], bl[13], 518, 544, 512, R, X16a, X16b, 512, nullptr);
    run_layer(Wl[14], bl[14], 512, 512, 512, R, X16b, X16a, 512, nullptr);
    run_layer(Wl[15], bl[15], 512, 512, 512, R, X16a, nullptr, 0, (float*)d_out);
  }

  // output tuple tail: pos passthrough
  hipMemcpyAsync((float*)d_out + (size_t)B*4096*512, pos,
                 (size_t)B*4096*3*sizeof(float), hipMemcpyDeviceToDevice, stream);
}
